// OutputPPBlockSMP_32384053412130
// MI455X (gfx1250) — matrix operand order repaired and checked by reading, not compiled, not run on hardware
//
#include <hip/hip_runtime.h>
#include <hip/hip_bf16.h>

// ---------------- constants from the reference ----------------
#define HC 128      // hidden_channels
#define DC 256      // out_emb_channels
#define NLAYER 3
#define NRAD 6
#define NBT 7       // 6 bond-type rows + 1 base row in W_rbf
#define ACT_W 256   // LDS activation row stride (elements)

typedef __attribute__((ext_vector_type(16))) __bf16 v16bf;
typedef __attribute__((ext_vector_type(8)))  __bf16 v8bf;
typedef __attribute__((ext_vector_type(8)))  float  v8f;
typedef __attribute__((ext_vector_type(4)))  float  f4v;

// ---------------------------------------------------------------------------
// Kernel 0: zero x_tot; convert weights to bf16 AND pre-swizzle them into
// fragment-major order so each wave's WMMA B-operand is a fully coalesced
// 1 KB contiguous chunk (32 B per lane):
//   W_up : s = ((nbI*4 + kbI)*32 + lane)*16 + j
//   Wls  : s = (((l*16 + nbI)*8 + kbI)*32 + lane)*16 + j
// where col = nbI*16 + (lane&15), k = kbI*32 + (lane>>4)*8 + (j&7) + (j>>3)*16.
// ---------------------------------------------------------------------------
__global__ void prep_kernel(const float* __restrict__ Wup,
                            const float* __restrict__ Wls,
                            __bf16* __restrict__ WupB,
                            __bf16* __restrict__ WlB,
                            float* __restrict__ xtot, int nXtot) {
  int s = blockIdx.x * blockDim.x + threadIdx.x;
  if (s < nXtot) xtot[s] = 0.0f;
  if (s < DC * HC) {                       // W_up swizzle: 16*4*32*16 = 32768
    int j = s & 15, lane = (s >> 4) & 31, kbI = (s >> 9) & 3, nbI = (s >> 11) & 15;
    int col = nbI * 16 + (lane & 15);
    int k = kbI * 32 + (lane >> 4) * 8 + (j & 7) + ((j >> 3) << 4);
    WupB[s] = (__bf16)Wup[col * HC + k];
  }
  if (s < NLAYER * DC * DC) {              // Wls swizzle: 3*16*8*32*16 = 196608
    int j = s & 15, lane = (s >> 4) & 31, kbI = (s >> 9) & 7;
    int nbI = (s >> 12) & 15, l = s >> 16;
    int col = nbI * 16 + (lane & 15);
    int k = kbI * 32 + (lane >> 4) * 8 + (j & 7) + ((j >> 3) << 4);
    WlB[s] = (__bf16)Wls[(l * DC + col) * DC + k];
  }
}

// ---------------------------------------------------------------------------
// Kernel 1: edge scatter.  8 threads per edge, 16 channels each.
// x is streamed with TH=NT so the 512 MB one-shot read does not evict the
// x_tot accumulator (25.6 MB, hot for 128M atomic adds) from L2.
// ---------------------------------------------------------------------------
__global__ __launch_bounds__(256) void scatter_kernel(
    const float* __restrict__ x, const float* __restrict__ rbf,
    const int* __restrict__ nodeIdx, const int* __restrict__ bt,
    const float* __restrict__ Wrbf, float* __restrict__ xtot, int E) {
  __shared__ float WL[NBT * HC * NRAD];   // 21504 B
  for (int t = threadIdx.x; t < NBT * HC * NRAD; t += 256) WL[t] = Wrbf[t];
  __syncthreads();

  const int part = threadIdx.x & 7;
  const int chBase = part << 4;           // 16 channels per thread
  const int e = blockIdx.x * 32 + (threadIdx.x >> 3);
  if (e >= E) return;

  const float r0 = rbf[e * 6 + 0], r1 = rbf[e * 6 + 1], r2 = rbf[e * 6 + 2];
  const float r3 = rbf[e * 6 + 3], r4 = rbf[e * 6 + 4], r5 = rbf[e * 6 + 5];
  const int node = nodeIdx[e];
  const int sel = bt[e] + 1;              // 0..6 ; 6 == "no bond type"
  const bool hasBond = (sel < 6);
  const float* WB = &WL[6 * HC * NRAD];
  const float* WS = hasBond ? &WL[sel * HC * NRAD] : WB;  // dummy row if unused
  const float wBond = hasBond ? 0.5f : 0.0f;              // (1 - ALPHA)

  float* dst = xtot + (size_t)node * HC;
  const float* xr = x + (size_t)e * HC;

#pragma unroll
  for (int cc = 0; cc < 16; cc += 4) {
    f4v xv = __builtin_nontemporal_load((const f4v*)(xr + chBase + cc));
#pragma unroll
    for (int q = 0; q < 4; ++q) {
      const int ch = chBase + cc + q;
      const float* wb = &WB[ch * NRAD];
      const float* ws = &WS[ch * NRAD];
      float g  = r0*wb[0] + r1*wb[1] + r2*wb[2] + r3*wb[3] + r4*wb[4] + r5*wb[5];
      float g2 = r0*ws[0] + r1*ws[1] + r2*ws[2] + r3*ws[3] + r4*ws[4] + r5*ws[5];
      float coef = 0.5f * g + wBond * g2;   // ALPHA = 0.5
      unsafeAtomicAdd(dst + ch, coef * xv[q]);
    }
  }
}

// ---------------------------------------------------------------------------
// WMMA fragment helpers.
// A (16x32, 16-bit, from LDS): lane row = lane%16; VGPR0-3 <- K[kb+8*half..+7],
//                              VGPR4-7 <- K[kb+16+8*half..+7].
// B (32x16, 16-bit, from global): pre-swizzled fragment-major; per lane 32
//                                 contiguous bytes at base + lane*32.
// ---------------------------------------------------------------------------
__device__ __forceinline__ v16bf frag_a(const __bf16* actRow, int kb, int half) {
  const __bf16* p0 = actRow + kb + 8 * half;
  v8bf lo = *(const v8bf*)p0;
  v8bf hi = *(const v8bf*)(p0 + 16);
  v16bf a;
#pragma unroll
  for (int j = 0; j < 8; ++j) { a[j] = lo[j]; a[j + 8] = hi[j]; }
  return a;
}

__device__ __forceinline__ v16bf frag_b(const __bf16* p) {
  return *(const v16bf*)p;   // 32 B, 32-B aligned, lane-coalesced
}

// ---------------------------------------------------------------------------
// Kernel 2: fused node MLP. 16 nodes per 512-thread workgroup (16 wave32).
// Wave w computes output channels [16w, 16w+16) of every layer.
// ---------------------------------------------------------------------------
__global__ __launch_bounds__(512) void mlp_kernel(
    const float* __restrict__ xtot,
    const __bf16* __restrict__ WupB, const __bf16* __restrict__ WlB,
    const float* __restrict__ bls, const float* __restrict__ Wout,
    float* __restrict__ out, int nNodes) {
  __shared__ __bf16 act[2][16][ACT_W];    // 16 KB ping-pong activations
  const int tid  = threadIdx.x;
  const int wid  = tid >> 5;
  const int lane = tid & 31;
  const int half = lane >> 4;
  const int rr   = lane & 15;             // A row within tile / B column
  const int nb   = wid << 4;              // this wave's output-channel base
  const int nodeBase = blockIdx.x << 4;

  // Stage x_tot tile (16 x 128 f32) into LDS as bf16.
  for (int idx = tid; idx < 16 * HC; idx += 512) {
    int r = idx >> 7, c = idx & (HC - 1);
    int node = nodeBase + r;
    float v = (node < nNodes) ? xtot[(size_t)node * HC + c] : 0.0f;
    act[0][r][c] = (__bf16)v;
  }
  __syncthreads();

  // ---- up layer: [16,128] @ W_up.T -> [16,256]  (no bias / no activation)
  {
    v8f acc = {};
    const __bf16* wbase = WupB + ((size_t)(wid * 4) * 32 + lane) * 16;
#pragma unroll
    for (int kbI = 0; kbI < 4; ++kbI) {
      v16bf a = frag_a(&act[0][rr][0], kbI * 32, half);
      v16bf b = frag_b(wbase + kbI * 512);
      acc = __builtin_amdgcn_wmma_f32_16x16x32_bf16(false, a, false, b,
                                                    (short)0, acc, false, false);
    }
    const int col = nb + rr;
#pragma unroll
    for (int v = 0; v < 8; ++v) act[1][v + 8 * half][col] = (__bf16)acc[v];
  }
  __syncthreads();

  // ---- 3 hidden layers: h = silu(h @ Wl.T + bl), K = 256
  int cur = 1;
#pragma unroll
  for (int l = 0; l < NLAYER; ++l) {
    const int col = nb + rr;
    const float bias = bls[l * DC + col];
    v8f acc = {};
    const __bf16* wbase = WlB + ((size_t)((l * 16 + wid) * 8) * 32 + lane) * 16;
#pragma unroll
    for (int kbI = 0; kbI < 8; ++kbI) {
      v16bf a = frag_a(&act[cur][rr][0], kbI * 32, half);
      v16bf b = frag_b(wbase + kbI * 512);
      acc = __builtin_amdgcn_wmma_f32_16x16x32_bf16(false, a, false, b,
                                                    (short)0, acc, false, false);
    }
    const int nxt = cur ^ 1;
#pragma unroll
    for (int v = 0; v < 8; ++v) {
      float h = acc[v] + bias;
      float s = h / (1.0f + __expf(-h));        // SiLU
      act[nxt][v + 8 * half][col] = (__bf16)s;
    }
    cur = nxt;
    __syncthreads();
  }

  // ---- final projection to C = 1
  if (tid < 16) {
    int node = nodeBase + tid;
    if (node < nNodes) {
      float acc = 0.0f;
#pragma unroll 8
      for (int k = 0; k < DC; ++k) acc += (float)act[cur][tid][k] * Wout[k];
      out[node] = acc;
    }
  }
}

// ---------------------------------------------------------------------------
// Host launcher
// ---------------------------------------------------------------------------
extern "C" void kernel_launch(void* const* d_in, const int* in_sizes, int n_in,
                              void* d_out, int out_size, void* d_ws, size_t ws_size,
                              hipStream_t stream) {
  const float* x    = (const float*)d_in[0];   // [E, 128]
  const float* rbf  = (const float*)d_in[1];   // [E, 6]
  const int*   iidx = (const int*)d_in[2];     // [E]
  // d_in[3] = num_nodes (device scalar; N recovered from out_size since C==1)
  const int*   bt   = (const int*)d_in[4];     // [E]
  const float* Wrbf = (const float*)d_in[5];   // [7, 128, 6]
  const float* Wup  = (const float*)d_in[6];   // [256, 128]
  const float* Wls  = (const float*)d_in[7];   // [3, 256, 256]
  const float* blsP = (const float*)d_in[8];   // [3, 256]
  const float* Wout = (const float*)d_in[9];   // [1, 256]

  const int E = in_sizes[2];
  const int N = out_size;                      // C == 1

  // workspace layout: x_tot f32 [N*128] | W_up bf16 (swizzled) | Wls bf16 (swizzled)
  char* ws = (char*)d_ws;
  float* xtot = (float*)ws;
  size_t off = ((size_t)N * HC * sizeof(float) + 63) & ~(size_t)63;
  __bf16* WupB = (__bf16*)(ws + off);
  __bf16* WlB  = (__bf16*)(ws + off + (size_t)DC * HC * sizeof(__bf16));

  const int nXtot = N * HC;
  {
    int threads = 256;
    int blocks = (nXtot + threads - 1) / threads;
    prep_kernel<<<blocks, threads, 0, stream>>>(Wup, Wls, WupB, WlB, xtot, nXtot);
  }
  {
    int blocks = (E + 31) / 32;                // 32 edges per 256-thread block
    scatter_kernel<<<blocks, 256, 0, stream>>>(x, rbf, iidx, bt, Wrbf, xtot, E);
  }
  {
    int blocks = (N + 15) / 16;                // 16 nodes per workgroup
    mlp_kernel<<<blocks, 512, 0, stream>>>(xtot, WupB, WlB, blsP, Wout,
                                           (float*)d_out, N);
  }
}
